// MultiheadAttention_9749575762416
// MI455X (gfx1250) — matrix operand order repaired and checked by reading, not compiled, not run on hardware
//
#include <hip/hip_runtime.h>
#include <hip/hip_bf16.h>
#include <cstdint>

// ---------------------------------------------------------------------------
// MI455X (gfx1250, wave32) MHA forward.
// GEMMs: v_wmma_f32_16x16x32_bf16 fed by double-buffered
// global_load_async_to_lds_b128 staging (ASYNCcnt-tracked).
// ---------------------------------------------------------------------------

typedef __bf16 v16bf __attribute__((ext_vector_type(16)));
typedef float  v8f   __attribute__((ext_vector_type(8)));

union FragU { uint4 u4[2]; v16bf v; };

__device__ __forceinline__ __bf16 f2bf(float x) {
    unsigned u = __float_as_uint(x);
    u += 0x7FFFu + ((u >> 16) & 1u);           // round-to-nearest-even
    unsigned short h = (unsigned short)(u >> 16);
    __bf16 r;
    __builtin_memcpy(&r, &h, 2);
    return r;
}

// Async 16B copy global -> LDS (per-lane addresses), tracked by ASYNCcnt.
__device__ __forceinline__ void async_copy_b128(unsigned lds_byte_off, const void* g) {
    asm volatile("global_load_async_to_lds_b128 %0, %1, off"
                 :: "v"(lds_byte_off), "v"((unsigned long long)(uintptr_t)g)
                 : "memory");
}

// LDS tile pitch: 40 halves (80 bytes) per row -> ds_load_b128 reads at
// row strides of 20 banks, conflict-free across 16 lanes, 16B-aligned.
#define LPITCH 40

// A fragment, 16x32 (MxK), 16-bit layout (ISA 7.12.2):
//   lanes 0-15 : M=lane,    halves = K {0..7, 16..23}
//   lanes16-31 : M=lane-16, halves = K {8..15, 24..31}
__device__ __forceinline__ v16bf lds_fragA(const __bf16* tile, int row0) {
    int lane = threadIdx.x & 31;
    int r    = row0 + (lane & 15);
    int koff = (lane >> 4) * 8;
    const __bf16* p = tile + r * LPITCH + koff;
    FragU f;
    f.u4[0] = *(const uint4*)(p);        // ds_load_b128
    f.u4[1] = *(const uint4*)(p + 16);
    return f.v;
}

// B fragment, 32x16 (KxN), operand stored [N,K] row-major:
//   lanes 0-15 : N=lane,    halves = K {0..7, 16..23}
//   lanes16-31 : N=lane-16, halves = K {8..15, 24..31}
__device__ __forceinline__ v16bf lds_fragB(const __bf16* tile, int row0) {
    int lane = threadIdx.x & 31;
    int n    = row0 + (lane & 15);
    int koff = (lane >> 4) * 8;
    const __bf16* p = tile + n * LPITCH + koff;
    FragU f;
    f.u4[0] = *(const uint4*)(p);
    f.u4[1] = *(const uint4*)(p + 16);
    return f.v;
}

// Cooperative async stage of one 128x32-half panel into LDS.
// 256 threads: thread t moves 16B chunks; row = t/4 (+64), kchunk = t%4.
__device__ __forceinline__ void stage_panel(const __bf16* gbase, int ld, int k0,
                                            __bf16* tile) {
    int t     = threadIdx.x;
    int chunk = (t & 3) * 8;             // halves
    int row   = t >> 2;                  // 0..63
#pragma unroll
    for (int p = 0; p < 2; ++p) {
        int r = row + p * 64;
        const __bf16* g = gbase + (size_t)r * ld + k0 + chunk;
        unsigned lds = (unsigned)(uintptr_t)(tile + r * LPITCH + chunk);
        async_copy_b128(lds, g);
    }
}

// Generic C[M,N] = A[M,K] * B[N,K]^T + bias[N]   (K-contiguous bf16 -> f32)
// Block: 256 threads = 8 waves; block tile 128x128; wave tile 64x32 (4x2 wmma).
__global__ __launch_bounds__(256)
void gemm_bf16_wmma(const __bf16* __restrict__ A, int lda,
                    const __bf16* __restrict__ B, int ldb,
                    float* __restrict__ C, int ldc,
                    const float* __restrict__ bias,
                    int M, int N, int K) {
    __shared__ __align__(16) __bf16 smA[2][128 * LPITCH];
    __shared__ __align__(16) __bf16 smB[2][128 * LPITCH];

    const int wave = threadIdx.x >> 5;
    const int wm   = wave >> 2;          // 0..1
    const int wn   = wave & 3;           // 0..3
    const int blockRow = blockIdx.y * 128;
    const int blockCol = blockIdx.x * 128;

    const __bf16* Ablk = A + (size_t)blockRow * lda;
    const __bf16* Bblk = B + (size_t)blockCol * ldb;

    v8f acc[4][2] = {};

    // Prologue: stage k0=0 into buffer 0 (4 async insts per wave).
    stage_panel(Ablk, lda, 0, smA[0]);
    stage_panel(Bblk, ldb, 0, smB[0]);

    for (int k0 = 0; k0 < K; k0 += 32) {
        const int cur = (k0 >> 5) & 1;
        if (k0 + 32 < K) {
            stage_panel(Ablk, lda, k0 + 32, smA[1 - cur]);
            stage_panel(Bblk, ldb, k0 + 32, smB[1 - cur]);
            // allow the 4 just-issued to remain outstanding; current buffer done
            asm volatile("s_wait_asynccnt 0x4" ::: "memory");
        } else {
            asm volatile("s_wait_asynccnt 0x0" ::: "memory");
        }
        __syncthreads();                 // current panels visible to all waves

        v16bf af[4];
#pragma unroll
        for (int i = 0; i < 4; ++i)
            af[i] = lds_fragA(smA[cur], wm * 64 + i * 16);
        v16bf bfm[2];
#pragma unroll
        for (int j = 0; j < 2; ++j)
            bfm[j] = lds_fragB(smB[cur], wn * 32 + j * 16);
#pragma unroll
        for (int i = 0; i < 4; ++i)
#pragma unroll
            for (int j = 0; j < 2; ++j)
                acc[i][j] = __builtin_amdgcn_wmma_f32_16x16x32_bf16(
                    false, af[i], false, bfm[j], (short)0, acc[i][j], false, false);

        __syncthreads();                 // all reads done before buffer is reused
    }

    // C layout: VGPR v -> M = v + 8*(lane>=16); N = lane&15
    const int lane = threadIdx.x & 31;
    const int cn   = lane & 15;
    const int rb   = (lane >> 4) * 8;
#pragma unroll
    for (int i = 0; i < 4; ++i) {
        int row0 = blockRow + wm * 64 + i * 16 + rb;
#pragma unroll
        for (int j = 0; j < 2; ++j) {
            int col = blockCol + wn * 32 + j * 16 + cn;
            if (col < N) {
                float bv = bias ? bias[col] : 0.0f;
#pragma unroll
                for (int v = 0; v < 8; ++v)
                    C[(size_t)(row0 + v) * ldc + col] = acc[i][j][v] + bv;
            }
        }
    }
}

// ---------------------------------------------------------------------------
__global__ void cvt_f32_bf16(const float* __restrict__ in, __bf16* __restrict__ out, int n) {
    int i = blockIdx.x * 256 + threadIdx.x;
    if (i < n) out[i] = f2bf(in[i]);
}

__global__ void zero_f32(float* __restrict__ p, int n) {
    int i = blockIdx.x * 256 + threadIdx.x;
    if (i < n) p[i] = 0.0f;
}

// qkv [T,B,3E] f32 -> q_bf[B*H,T,hd] (scaled), k_bf[B*H,T,hd], vT_bf[B*H,hd,T]
__global__ void split_qkv(const float* __restrict__ qkv,
                          __bf16* __restrict__ q, __bf16* __restrict__ k,
                          __bf16* __restrict__ vT,
                          int T, int B, int E, int H, float scaling) {
    int idx = blockIdx.x * 256 + threadIdx.x;
    int n = T * B * 3 * E;
    if (idx >= n) return;
    int hd = E / H;
    int f  = idx % (3 * E);
    int tb = idx / (3 * E);
    int b  = tb % B;
    int t  = tb / B;
    int sec = f / E;
    int e   = f % E;
    int h   = e / hd;
    int d   = e % hd;
    size_t head = (size_t)(b * H + h);
    float x = qkv[idx];
    if (sec == 0)      q[(head * T + t) * hd + d] = f2bf(x * scaling);
    else if (sec == 1) k[(head * T + t) * hd + d] = f2bf(x);
    else               vT[head * (size_t)hd * T + (size_t)d * T + t] = f2bf(x);
}

// One block per row: softmax of scores[row,:], write bf16 probs, avg += p/H
__global__ __launch_bounds__(256)
void softmax_row(const float* __restrict__ S, __bf16* __restrict__ P,
                 float* __restrict__ avg, int T, float invH) {
    __shared__ float red[256];
    const int row = blockIdx.x;
    const int tid = threadIdx.x;
    const float* s = S + (size_t)row * T;

    float e[8];
    float m = -3.4e38f;
#pragma unroll
    for (int i = 0; i < 8; ++i) { e[i] = s[tid + i * 256]; m = fmaxf(m, e[i]); }
    red[tid] = m; __syncthreads();
    for (int off = 128; off > 0; off >>= 1) {
        if (tid < off) red[tid] = fmaxf(red[tid], red[tid + off]);
        __syncthreads();
    }
    m = red[0]; __syncthreads();

    float sum = 0.0f;
#pragma unroll
    for (int i = 0; i < 8; ++i) { e[i] = __expf(e[i] - m); sum += e[i]; }
    red[tid] = sum; __syncthreads();
    for (int off = 128; off > 0; off >>= 1) {
        if (tid < off) red[tid] += red[tid + off];
        __syncthreads();
    }
    float r = 1.0f / red[0];

#pragma unroll
    for (int i = 0; i < 8; ++i) {
        int c = tid + i * 256;
        float p = e[i] * r;
        P[(size_t)row * T + c] = f2bf(p);
        avg[(size_t)row * T + c] += p * invH;
    }
}

// ---------------------------------------------------------------------------
extern "C" void kernel_launch(void* const* d_in, const int* in_sizes, int n_in,
                              void* d_out, int out_size, void* d_ws, size_t ws_size,
                              hipStream_t stream) {
    const int T = 2048, Bb = 2, E = 1024, H = 16, hd = 64;
    const int TB = T * Bb;                  // 4096 GEMM rows
    const float scaling = 0.125f;           // hd^-0.5

    const float* query = (const float*)d_in[0];
    const float* W_qkv = (const float*)d_in[1];
    const float* b_qkv = (const float*)d_in[2];
    const float* W_o   = (const float*)d_in[3];
    const float* b_o   = (const float*)d_in[4];

    char* w = (char*)d_ws;
    size_t off = 0;
    auto alloc = [&](size_t bytes) {
        void* p = w + off;
        off = (off + bytes + 255) & ~(size_t)255;
        return p;
    };
    __bf16* query_bf = (__bf16*)alloc((size_t)TB * E * 2);
    __bf16* Wqkv_bf  = (__bf16*)alloc((size_t)3 * E * E * 2);
    __bf16* Wo_bf    = (__bf16*)alloc((size_t)E * E * 2);
    __bf16* q_bf     = (__bf16*)alloc((size_t)Bb * H * T * hd * 2);
    __bf16* k_bf     = (__bf16*)alloc((size_t)Bb * H * T * hd * 2);
    __bf16* vT_bf    = (__bf16*)alloc((size_t)Bb * H * hd * T * 2 + (size_t)hd * T * 2); // +1 head slack
    float*  qkv      = (float*)alloc((size_t)TB * 3 * E * 4);
    float*  scores   = (float*)alloc((size_t)T * T * 4);
    __bf16* attn_bf  = (__bf16*)alloc((size_t)T * T * 2);
    float*  ctx      = (float*)alloc((size_t)TB * E * 4);
    __bf16* ctx_bf   = (__bf16*)alloc((size_t)TB * E * 2);
    (void)ws_size; (void)in_sizes; (void)n_in; (void)out_size;

    float* out_p = (float*)d_out;                       // [T,B,E]
    float* avg_p = out_p + (size_t)TB * E;              // [B,T,T]

    auto g1 = [](int n) { return dim3((n + 255) / 256); };

    // 1) bf16 conversions of query and weights
    cvt_f32_bf16<<<g1(TB * E), 256, 0, stream>>>(query, query_bf, TB * E);
    cvt_f32_bf16<<<g1(3 * E * E), 256, 0, stream>>>(W_qkv, Wqkv_bf, 3 * E * E);
    cvt_f32_bf16<<<g1(E * E), 256, 0, stream>>>(W_o, Wo_bf, E * E);

    // 2) zero the avg-weights output region (d_out is poisoned)
    zero_f32<<<g1(Bb * T * T), 256, 0, stream>>>(avg_p, Bb * T * T);

    // 3) fused QKV projection: qkv[4096,3072] = query * W_qkv^T + b_qkv
    gemm_bf16_wmma<<<dim3(3 * E / 128, TB / 128), 256, 0, stream>>>(
        query_bf, E, Wqkv_bf, E, qkv, 3 * E, b_qkv, TB, 3 * E, E);

    // 4) split into heads (q scaled, v transposed)
    split_qkv<<<g1(TB * 3 * E), 256, 0, stream>>>(qkv, q_bf, k_bf, vT_bf,
                                                  T, Bb, E, H, scaling);

    // 5) per-head attention (sequential -> deterministic avg accumulation)
    for (int head = 0; head < Bb * H; ++head) {
        int b = head / H, h = head % H;
        const __bf16* qh = q_bf + (size_t)head * T * hd;
        const __bf16* kh = k_bf + (size_t)head * T * hd;
        const __bf16* vh = vT_bf + (size_t)head * hd * T;

        // scores[T,T] = q_h * k_h^T
        gemm_bf16_wmma<<<dim3(T / 128, T / 128), 256, 0, stream>>>(
            qh, hd, kh, hd, scores, T, nullptr, T, T, hd);

        // softmax rows -> bf16 probs; avg[b] += p/H
        softmax_row<<<dim3(T), 256, 0, stream>>>(
            scores, attn_bf, avg_p + (size_t)b * T * T, T, 1.0f / H);

        // ctx[:,head slice] = attn * v_h   (B operand = v^T[hd,T], K-contiguous)
        gemm_bf16_wmma<<<dim3(1, T / 128), 256, 0, stream>>>(
            attn_bf, T, vh, T, ctx + (size_t)b * E + (size_t)h * hd, Bb * E,
            nullptr, T, hd, T);
    }

    // 6) output projection: out[4096,1024] = ctx * W_o^T + b_o
    cvt_f32_bf16<<<g1(TB * E), 256, 0, stream>>>(ctx, ctx_bf, TB * E);
    gemm_bf16_wmma<<<dim3(E / 128, TB / 128), 256, 0, stream>>>(
        ctx_bf, E, Wo_bf, E, out_p, E, b_o, TB, E, E);
}
